// FlashAttention_29300266893601
// MI455X (gfx1250) — compile-verified, matrix operand order checked by reading, not run on hardware
//
#include <hip/hip_runtime.h>

typedef __attribute__((ext_vector_type(16))) _Float16 v16h;
typedef __attribute__((ext_vector_type(8)))  _Float16 v8h;
typedef __attribute__((ext_vector_type(8)))  float    v8f;
typedef __attribute__((ext_vector_type(4)))  float    v4f;

static constexpr int B_  = 2;
static constexpr int S_  = 2048;
static constexpr int D_  = 1024;
static constexpr int H_  = 16;
static constexpr int HD_ = 64;
#define ATT_SCALE 0.125f  /* HD^-0.5 */

__device__ __forceinline__ v8f wmma16(v16h a, v16h b, v8f c) {
  // D = A(16x32 f16) * B(32x16 f16) + C(16x16 f32)
  return __builtin_amdgcn_wmma_f32_16x16x32_f16(false, a, false, b, (short)0, c,
                                                false, false);
}

__device__ __forceinline__ v8h ld8(const _Float16* __restrict__ p) {
  return *(const v8h*)p;
}

__device__ __forceinline__ v16h cat8(v8h lo, v8h hi) {
  return __builtin_shufflevector(lo, hi, 0, 1, 2, 3, 4, 5, 6, 7,
                                 8, 9, 10, 11, 12, 13, 14, 15);
}

// generic pointer into __shared__ -> 32-bit LDS byte offset (LDS aperture
// occupies the high dword; addr[31:0] is the LDS offset per CDNA5 aperture map)
__device__ __forceinline__ uint32_t lds_off32(const void* p) {
  return (uint32_t)(uintptr_t)p;
}

// async global -> LDS copy, 16B per lane, tracked by ASYNCcnt (CDNA5)
__device__ __forceinline__ void async_b128(uint32_t lds_byte, uint64_t gaddr) {
  asm volatile("global_load_async_to_lds_b128 %0, %1, off"
               :: "v"(lds_byte), "v"(gaddr)
               : "memory");
}

// ---------------------------------------------------------------------------
// One-shot f32 -> f16 cast (8 elements / thread, vectorized 16B in / 16B out)
// ---------------------------------------------------------------------------
__global__ void __launch_bounds__(256)
cast_f16_kernel(const float* __restrict__ in, _Float16* __restrict__ out, int n) {
  const int i = (blockIdx.x * 256 + threadIdx.x) * 8;
  if (i >= n) return;
  v4f a = *(const v4f*)(in + i);
  v4f b = *(const v4f*)(in + i + 4);
  v8h r;
  r[0] = (_Float16)a[0]; r[1] = (_Float16)a[1];
  r[2] = (_Float16)a[2]; r[3] = (_Float16)a[3];
  r[4] = (_Float16)b[0]; r[5] = (_Float16)b[1];
  r[6] = (_Float16)b[2]; r[7] = (_Float16)b[3];
  *(v8h*)(out + i) = r;
}

// ---------------------------------------------------------------------------
// QKV projection: out = X @ W^T (all f16 in, f32 accumulate).
//   mode 0: Q  -> [B,H,S,HD], pre-scaled by ATT_SCALE
//   mode 1: K  -> [B,H,S,HD]
//   mode 2: V  -> [B,H,HD,S]  (transposed per head, contiguous 16B stores)
// One wave computes a 32x64 tile (8 WMMAs / K-step). 512 blocks x 128 threads.
// ---------------------------------------------------------------------------
__global__ void __launch_bounds__(128)
qkv_proj_kernel(const _Float16* __restrict__ X, const _Float16* __restrict__ W,
                _Float16* __restrict__ out, int mode) {
  const int lane   = threadIdx.x & 31;
  const int warp   = threadIdx.x >> 5;
  const int gw     = blockIdx.x * 4 + warp;   // 0..2047
  const int mTile  = gw >> 4;                 // 0..127 (32 rows each)
  const int nTile  = gw & 15;                 // 0..15  (64 cols each)
  const int lane16 = lane & 15;
  const int base8  = (lane >> 4) * 8;         // A/B frag K sub-offset

  const _Float16* x0 = X + (size_t)(mTile * 32 + lane16) * D_;
  const _Float16* x1 = x0 + (size_t)16 * D_;
  const _Float16* wp[4];
#pragma unroll
  for (int s = 0; s < 4; ++s)
    wp[s] = W + (size_t)(nTile * 64 + s * 16 + lane16) * D_;

  v8f c[8] = {};  // c[r*4 + s]: row-group r (0..1), col-group s (0..3)
  for (int k = 0; k < D_; k += 32) {
    const int ka = k + base8;
    v16h a0 = cat8(ld8(x0 + ka), ld8(x0 + ka + 16));
    v16h a1 = cat8(ld8(x1 + ka), ld8(x1 + ka + 16));
#pragma unroll
    for (int s = 0; s < 4; ++s) {
      v16h b = cat8(ld8(wp[s] + ka), ld8(wp[s] + ka + 16));
      c[s]     = wmma16(a0, b, c[s]);
      c[4 + s] = wmma16(a1, b, c[4 + s]);
    }
  }

  const float sc = (mode == 0) ? ATT_SCALE : 1.0f;
#pragma unroll
  for (int r = 0; r < 2; ++r) {
    const int M0 = mTile * 32 + r * 16 + base8;  // first row of half-group
    const int b  = M0 / S_;
    const int s0 = M0 % S_;
#pragma unroll
    for (int sub = 0; sub < 4; ++sub) {
      const int ncol = nTile * 64 + sub * 16 + lane16;
      const int h = ncol >> 6;
      const int d = ncol & 63;
      v8f cc = c[r * 4 + sub];
      if (mode < 2) {
        _Float16* o = out + ((size_t)(b * H_ + h) * S_ + s0) * HD_ + d;
#pragma unroll
        for (int j = 0; j < 8; ++j) o[(size_t)j * HD_] = (_Float16)(cc[j] * sc);
      } else {
        v8h o8;
#pragma unroll
        for (int j = 0; j < 8; ++j) o8[j] = (_Float16)cc[j];
        *(v8h*)(out + ((size_t)(b * H_ + h) * HD_ + d) * S_ + s0) = o8;
      }
    }
  }
}

// ---------------------------------------------------------------------------
// Causal flash attention, async-LDS double buffered.
// One wave per (b,h, 16-query tile); 32 keys per step. Per step:
//   - 16x global_load_async_to_lds_b128 stream the NEXT K/V tile into LDS
//   - s_wait_asynccnt fences the CURRENT tile
//   - 4 WMMAs QK^T, online softmax, P transposed via LDS, 4 WMMAs PV
// ---------------------------------------------------------------------------
__global__ void __launch_bounds__(64)
flash_attn_kernel(const _Float16* __restrict__ Q, const _Float16* __restrict__ K,
                  const _Float16* __restrict__ Vt, _Float16* __restrict__ AO) {
  __shared__ _Float16 ldsK[2][2][32 * 64];  // [wave][buf][key][hd]
  __shared__ _Float16 ldsV[2][2][64 * 32];  // [wave][buf][hd][key]
  __shared__ _Float16 plds[2][16 * 40];     // per-wave P tile, padded rows

  const int lane   = threadIdx.x & 31;
  const int warp   = threadIdx.x >> 5;
  const int gw     = blockIdx.x * 2 + warp;  // 0..4095
  const int qt     = gw & 127;               // query tile in sequence
  const int bh     = gw >> 7;                // 0..31 = b*H + h
  const int qbase  = qt * 16;
  const int lane16 = lane & 15;
  const int base8  = (lane >> 4) * 8;

  const _Float16* kbh = K  + (size_t)bh * S_ * HD_;
  const _Float16* vbh = Vt + (size_t)bh * HD_ * S_;

  // Q tile as two A-fragments (HD split 0..31 / 32..63)
  const _Float16* qp = Q + ((size_t)bh * S_ + qbase + lane16) * HD_;
  const v16h qa0 = cat8(ld8(qp + base8),      ld8(qp + base8 + 16));
  const v16h qa1 = cat8(ld8(qp + base8 + 32), ld8(qp + base8 + 48));

  v8f o0 = {}, o1 = {}, o2 = {}, o3 = {};
  float mrow[8], lrow[8];
#pragma unroll
  for (int j = 0; j < 8; ++j) { mrow[j] = -3.0e38f; lrow[j] = 0.0f; }

  // LDS byte offsets for this wave's staging buffers
  uint32_t klds[2], vlds[2];
  klds[0] = lds_off32(&ldsK[warp][0][0]);
  klds[1] = lds_off32(&ldsK[warp][1][0]);
  vlds[0] = lds_off32(&ldsV[warp][0][0]);
  vlds[1] = lds_off32(&ldsV[warp][1][0]);
  _Float16* pl = plds[warp];

  // ---- issue one K/V tile (32 keys) as 16 async b128 copies -------------
  auto issue_tile = [&](int kb, int buf) {
    // K tile: contiguous 4KB starting at key kb (rows of [S,HD])
    const uint64_t kg = (uint64_t)(uintptr_t)(kbh + (size_t)kb * HD_) +
                        (uint64_t)lane * 16;
    const uint32_t kl = klds[buf] + (uint32_t)lane * 16;
#pragma unroll
    for (int i = 0; i < 8; ++i) async_b128(kl + i * 512, kg + i * 512);
    // V^T tile: 64 rows x 64B, global row stride S_ halfs -> LDS rows of 32
#pragma unroll
    for (int i = 0; i < 8; ++i) {
      const int ch = i * 32 + lane;   // 16B chunk id, 256 total
      const int d  = ch >> 2;         // hd row 0..63
      const int sb = ch & 3;          // 16B sub-chunk in row
      const uint64_t vg =
          (uint64_t)(uintptr_t)(vbh + (size_t)d * S_ + kb + sb * 8);
      const uint32_t vl = vlds[buf] + (uint32_t)(d * 32 + sb * 8) * 2;
      async_b128(vl, vg);
    }
  };

  issue_tile(0, 0);
  int cur = 0;

  for (int kb = 0; kb <= qbase; kb += 32) {
    if (kb + 32 <= qbase) {
      issue_tile(kb + 32, cur ^ 1);
      asm volatile("s_wait_asynccnt 16" ::: "memory");  // current tile landed
    } else {
      asm volatile("s_wait_asynccnt 0" ::: "memory");
    }

    const _Float16* kc = (const _Float16*)&ldsK[warp][cur][0];
    const _Float16* vc = (const _Float16*)&ldsV[warp][cur][0];

    // K B-fragments from LDS (ds_load_b128)
    const _Float16* kp0 = kc + lane16 * HD_;
    const _Float16* kp1 = kc + (16 + lane16) * HD_;
    v16h k00 = cat8(ld8(kp0 + base8),      ld8(kp0 + base8 + 16));
    v16h k01 = cat8(ld8(kp0 + base8 + 32), ld8(kp0 + base8 + 48));
    v16h k10 = cat8(ld8(kp1 + base8),      ld8(kp1 + base8 + 16));
    v16h k11 = cat8(ld8(kp1 + base8 + 32), ld8(kp1 + base8 + 48));

    v8f s0 = {}, s1 = {};
    s0 = wmma16(qa0, k00, s0);
    s0 = wmma16(qa1, k01, s0);
    s1 = wmma16(qa0, k10, s1);
    s1 = wmma16(qa1, k11, s1);

    // causal mask (only tiles touching the diagonal need it)
    if (kb + 15 > qbase) {
#pragma unroll
      for (int j = 0; j < 8; ++j)
        if (kb + lane16 > qbase + base8 + j) s0[j] = -1.0e9f;
    }
    if (kb + 31 > qbase) {
#pragma unroll
      for (int j = 0; j < 8; ++j)
        if (kb + 16 + lane16 > qbase + base8 + j) s1[j] = -1.0e9f;
    }

    // online softmax; each row lives in one 16-lane half-group
#pragma unroll
    for (int j = 0; j < 8; ++j) {
      float t = fmaxf(s0[j], s1[j]);
      t = fmaxf(t, __shfl_xor(t, 1));
      t = fmaxf(t, __shfl_xor(t, 2));
      t = fmaxf(t, __shfl_xor(t, 4));
      t = fmaxf(t, __shfl_xor(t, 8));
      const float mn   = fmaxf(mrow[j], t);
      const float corr = __expf(mrow[j] - mn);
      mrow[j] = mn;
      o0[j] *= corr; o1[j] *= corr; o2[j] *= corr; o3[j] *= corr;
      const float p0 = __expf(s0[j] - mn);
      const float p1 = __expf(s1[j] - mn);
      s0[j] = p0; s1[j] = p1;
      float rs = p0 + p1;
      rs += __shfl_xor(rs, 1);
      rs += __shfl_xor(rs, 2);
      rs += __shfl_xor(rs, 4);
      rs += __shfl_xor(rs, 8);
      lrow[j] = lrow[j] * corr + rs;
    }

    // C-layout P -> LDS -> A-layout fragment (16x32, K = 32 keys)
#pragma unroll
    for (int j = 0; j < 8; ++j) {
      pl[(base8 + j) * 40 + lane16]      = (_Float16)s0[j];
      pl[(base8 + j) * 40 + 16 + lane16] = (_Float16)s1[j];
    }
    const _Float16* pr = pl + lane16 * 40;
    v16h pa = cat8(ld8(pr + base8), ld8(pr + base8 + 16));

    // V B-fragments from LDS (rows of 32 keys)
    const _Float16* vpA = vc + lane16 * 32;
    const _Float16* vpB = vpA + 16 * 32;
    const _Float16* vpC = vpA + 32 * 32;
    const _Float16* vpD = vpA + 48 * 32;
    v16h vb0 = cat8(ld8(vpA + base8), ld8(vpA + base8 + 16));
    v16h vb1 = cat8(ld8(vpB + base8), ld8(vpB + base8 + 16));
    v16h vb2 = cat8(ld8(vpC + base8), ld8(vpC + base8 + 16));
    v16h vb3 = cat8(ld8(vpD + base8), ld8(vpD + base8 + 16));
    o0 = wmma16(pa, vb0, o0);
    o1 = wmma16(pa, vb1, o1);
    o2 = wmma16(pa, vb2, o2);
    o3 = wmma16(pa, vb3, o3);

    cur ^= 1;
  }

  const int b = bh >> 4;
  const int h = bh & 15;
  float inv[8];
#pragma unroll
  for (int j = 0; j < 8; ++j) inv[j] = 1.0f / lrow[j];

  v8f os[4] = {o0, o1, o2, o3};
#pragma unroll
  for (int nc = 0; nc < 4; ++nc) {
    const int col = h * HD_ + nc * 16 + lane16;
    _Float16* o = AO + ((size_t)b * S_ + qbase + base8) * D_ + col;
#pragma unroll
    for (int j = 0; j < 8; ++j) o[(size_t)j * D_] = (_Float16)(os[nc][j] * inv[j]);
  }
}

// ---------------------------------------------------------------------------
// Output projection: out(f32) = AO(f16) @ Wo^T(f16).  32x64 tile per wave.
// ---------------------------------------------------------------------------
__global__ void __launch_bounds__(128)
out_proj_kernel(const _Float16* __restrict__ A, const _Float16* __restrict__ W,
                float* __restrict__ out) {
  const int lane   = threadIdx.x & 31;
  const int warp   = threadIdx.x >> 5;
  const int gw     = blockIdx.x * 4 + warp;   // 0..2047
  const int mTile  = gw >> 4;
  const int nTile  = gw & 15;
  const int lane16 = lane & 15;
  const int base8  = (lane >> 4) * 8;

  const _Float16* a0p = A + (size_t)(mTile * 32 + lane16) * D_;
  const _Float16* a1p = a0p + (size_t)16 * D_;
  const _Float16* wp[4];
#pragma unroll
  for (int s = 0; s < 4; ++s)
    wp[s] = W + (size_t)(nTile * 64 + s * 16 + lane16) * D_;

  v8f c[8] = {};
  for (int k = 0; k < D_; k += 32) {
    const int ka = k + base8;
    v16h a0 = cat8(ld8(a0p + ka), ld8(a0p + ka + 16));
    v16h a1 = cat8(ld8(a1p + ka), ld8(a1p + ka + 16));
#pragma unroll
    for (int s = 0; s < 4; ++s) {
      v16h b = cat8(ld8(wp[s] + ka), ld8(wp[s] + ka + 16));
      c[s]     = wmma16(a0, b, c[s]);
      c[4 + s] = wmma16(a1, b, c[4 + s]);
    }
  }

#pragma unroll
  for (int r = 0; r < 2; ++r) {
    const int M0 = mTile * 32 + r * 16 + base8;
#pragma unroll
    for (int sub = 0; sub < 4; ++sub) {
      const int ncol = nTile * 64 + sub * 16 + lane16;
      float* o = out + (size_t)M0 * D_ + ncol;
#pragma unroll
      for (int j = 0; j < 8; ++j) o[(size_t)j * D_] = c[r * 4 + sub][j];
    }
  }
}

// ---------------------------------------------------------------------------
extern "C" void kernel_launch(void* const* d_in, const int* in_sizes, int n_in,
                              void* d_out, int out_size, void* d_ws, size_t ws_size,
                              hipStream_t stream) {
  (void)in_sizes; (void)n_in; (void)out_size; (void)ws_size;
  const float* query = (const float*)d_in[0];
  const float* key   = (const float*)d_in[1];
  const float* value = (const float*)d_in[2];
  const float* Wq    = (const float*)d_in[3];
  const float* Wk    = (const float*)d_in[4];
  const float* Wv    = (const float*)d_in[5];
  const float* Wo    = (const float*)d_in[6];

  _Float16* ws = (_Float16*)d_ws;
  const size_t NX = (size_t)B_ * S_ * D_;  // 4,194,304 elements
  const size_t NW = (size_t)D_ * D_;       // 1,048,576 elements
  _Float16* Xq  = ws;                      // aliased by AO after Q projection
  _Float16* Xk  = ws + NX;
  _Float16* Xv  = ws + 2 * NX;
  _Float16* W16q = ws + 3 * NX;
  _Float16* W16k = W16q + NW;
  _Float16* W16v = W16k + NW;
  _Float16* W16o = W16v + NW;
  _Float16* Qf  = W16o + NW;
  _Float16* Kf  = Qf + NX;
  _Float16* Vt  = Kf + NX;
  _Float16* AO  = Xq;  // query activations are dead once Q is projected

  const dim3 blk64(64);
  const dim3 blk128(128);
  const dim3 blk256(256);
  const dim3 gX((unsigned)(NX / 8 / 256));   // 2048 blocks
  const dim3 gW((unsigned)(NW / 8 / 256));   // 512 blocks
  const dim3 gGemm(512);                     // 2048 waves: 128 M x 16 N tiles
  const dim3 gAttn(2048);                    // 4096 waves: 32 bh x 128 q-tiles

  cast_f16_kernel<<<gX, blk256, 0, stream>>>(query, Xq, (int)NX);
  cast_f16_kernel<<<gX, blk256, 0, stream>>>(key,   Xk, (int)NX);
  cast_f16_kernel<<<gX, blk256, 0, stream>>>(value, Xv, (int)NX);
  cast_f16_kernel<<<gW, blk256, 0, stream>>>(Wq, W16q, (int)NW);
  cast_f16_kernel<<<gW, blk256, 0, stream>>>(Wk, W16k, (int)NW);
  cast_f16_kernel<<<gW, blk256, 0, stream>>>(Wv, W16v, (int)NW);
  cast_f16_kernel<<<gW, blk256, 0, stream>>>(Wo, W16o, (int)NW);

  qkv_proj_kernel<<<gGemm, blk128, 0, stream>>>(Xq, W16q, Qf, 0);
  qkv_proj_kernel<<<gGemm, blk128, 0, stream>>>(Xk, W16k, Kf, 1);
  qkv_proj_kernel<<<gGemm, blk128, 0, stream>>>(Xv, W16v, Vt, 2);
  flash_attn_kernel<<<gAttn, blk64, 0, stream>>>(Qf, Kf, Vt, AO);
  out_proj_kernel<<<gGemm, blk128, 0, stream>>>(AO, W16o, (float*)d_out);
}
